// SimpleTransformer_17712445129457
// MI455X (gfx1250) — matrix operand order repaired and checked by reading, not compiled, not run on hardware
//
#include <hip/hip_runtime.h>

// ---------------------------------------------------------------------------
// MI455X (gfx1250) transformer forward.  Compute-bound: ~360 GFLOP of GEMM vs
// <100MB resident (fits 192MB L2).  All matmuls via v_wmma_f32_16x16x32_f16
// with f32 accumulation; residual stream kept f32; weights converted to f16
// once per launch inside the workspace.  A-tiles are staged into LDS by the
// CDNA5 Tensor Data Mover (tensor_load_to_lds + s_wait_tensorcnt) when the
// toolchain exposes the builtin; B-tiles are transposed on store to match the
// WMMA B-fragment layout.
// ---------------------------------------------------------------------------

#define LSEQ   2048
#define BATCH  4
#define DMODEL 512
#define NHEADS 8
#define DK     64
#define DFF    2048
#define NROWS  (BATCH * LSEQ)   // 8192

typedef __attribute__((ext_vector_type(16))) _Float16     v16h;
typedef __attribute__((ext_vector_type(8)))  float        v8f;
typedef __attribute__((ext_vector_type(4)))  int          v4i;
typedef __attribute__((ext_vector_type(8)))  int          v8i;
typedef __attribute__((ext_vector_type(4)))  unsigned int v4u;

union U16h { v16h v; v4i i[2]; _Float16 h[16]; };
union U8h  { v4i  v; _Float16 h[8]; };

// A-fragment (16x32 f16): lane holds 8 halves at p[0..7] and 8 at p[16..23],
// where p is already offset by (lane>=16 ? 8 : 0) in K.
static __device__ inline v16h ldA_frag(const _Float16* p) {
  U16h u; u.i[0] = *(const v4i*)p; u.i[1] = *(const v4i*)(p + 16); return u.v;
}
// B-fragment (32x16 f16): lane holds 8 halves at p[0..7] and 8 at p[16..23],
// where p is already offset by (lane>=16 ? 8 : 0) in K.
static __device__ inline v16h ldB_frag(const _Float16* p) {
  U16h u; u.i[0] = *(const v4i*)p; u.i[1] = *(const v4i*)(p + 16); return u.v;
}

static __device__ inline v8f wmma_f16(v16h a, v16h b, v8f c) {
  return __builtin_amdgcn_wmma_f32_16x16x32_f16(false, a, false, b, (short)0, c,
                                                false, false);
}

// ---------------------------------------------------------------------------
// Tensor Data Mover: DMA a 2D tile of f16 (tile_w x tile_h, row stride in
// elements) from global memory into LDS at lds_byte_off.  D# bitfields per
// CDNA5 ISA ch.8: group0 = {count=1, lds_addr, global_addr[56:0], type=2},
// group1 = {data_size=2B, tensor_dim0/1, tile_dim0/1, tensor_dim0_stride}.
// Wave-level op (EXEC ignored): issue from one wave per workgroup.
// ---------------------------------------------------------------------------
#if defined(__HIP_DEVICE_COMPILE__) &&                       \
    __has_builtin(__builtin_amdgcn_tensor_load_to_lds) &&    \
    __has_builtin(__builtin_amdgcn_s_wait_tensorcnt)
#define HAVE_TDM 1
static __device__ inline void tdm_load_tile_2d(unsigned int lds_byte_off,
                                               const void* gaddr,
                                               unsigned int tile_w,
                                               unsigned int tile_h,
                                               unsigned int tensor_w,
                                               unsigned int tensor_h,
                                               unsigned long long row_stride) {
  unsigned long long ga = (unsigned long long)(uintptr_t)gaddr;
  v4u g0 = {0u, 0u, 0u, 0u};
  g0.x = 1u;                                              // count=1 (valid)
  g0.y = lds_byte_off;                                    // LDS dest (bytes)
  g0.z = (unsigned int)ga;                                // global addr lo
  g0.w = ((unsigned int)(ga >> 32) & 0x01FFFFFFu) | (2u << 30);  // hi + type=2
  v8i g1 = {0, 0, 0, 0, 0, 0, 0, 0};
  g1[0] = (int)(1u << 16);                                // data_size = 2 bytes
  g1[1] = (int)((tensor_w & 0xFFFFu) << 16);              // tensor_dim0[15:0]
  g1[2] = (int)((tensor_w >> 16) | ((tensor_h & 0xFFFFu) << 16));
  g1[3] = (int)((tensor_h >> 16) | (tile_w << 16));       // tile_dim0
  g1[4] = (int)(tile_h & 0xFFFFu);                        // tile_dim1
  g1[5] = (int)(unsigned int)row_stride;                  // dim0 stride lo
  g1[6] = (int)((unsigned int)(row_stride >> 32) & 0xFFFFu);
  v4i g2 = {0, 0, 0, 0};
  v4i g3 = {0, 0, 0, 0};
#if __clang_major__ >= 23
  v8i g4 = {0, 0, 0, 0, 0, 0, 0, 0};
  __builtin_amdgcn_tensor_load_to_lds(g0, g1, g2, g3, g4, 0);
#else
  __builtin_amdgcn_tensor_load_to_lds(g0, g1, g2, g3, 0);
#endif
}
#endif

// ---------------------------------------------------------------------------
// f32 -> f16 conversion (weights)
// ---------------------------------------------------------------------------
__global__ void cvt_f32_f16(const float* __restrict__ src,
                            _Float16* __restrict__ dst, int n) {
  int i = blockIdx.x * blockDim.x + threadIdx.x;
  if (i < n) dst[i] = (_Float16)src[i];
}

// ---------------------------------------------------------------------------
// per-(d,t) mean over batch of observed values
// ---------------------------------------------------------------------------
__global__ void posmean_kernel(const float* __restrict__ X,
                               const float* __restrict__ Mk,
                               float* __restrict__ pm) {
  int i = blockIdx.x * blockDim.x + threadIdx.x;
  if (i >= LSEQ) return;
  float s = 0.f, c = 0.f;
  for (int b = 0; b < BATCH; ++b) {
    float m = Mk[b * LSEQ + i];
    s += X[b * LSEQ + i] * m;
    c += m;
  }
  pm[i] = s / (c + 1e-10f);
}

// ---------------------------------------------------------------------------
// impute + input projection (fan_in=2) + positional encoding  -> h (f32)
// ---------------------------------------------------------------------------
__global__ void embed_kernel(const float* __restrict__ X,
                             const float* __restrict__ Mk,
                             const float* __restrict__ pm,
                             const float* __restrict__ Wi,   // (2,512)
                             const float* __restrict__ bi,
                             const float* __restrict__ pe,   // (2048,512)
                             float* __restrict__ h) {
  int row = blockIdx.x;            // b*LSEQ + l
  int l = row & (LSEQ - 1);
  float m  = Mk[row];
  float xv = X[row] * m + (1.f - m) * pm[l];
  for (int c = threadIdx.x; c < DMODEL; c += blockDim.x)
    h[(size_t)row * DMODEL + c] =
        Wi[c] * xv + Wi[DMODEL + c] * m + bi[c] + pe[(size_t)l * DMODEL + c];
}

// ---------------------------------------------------------------------------
// LayerNorm, one wave (32 lanes) per row of 512.
// ---------------------------------------------------------------------------
__global__ __launch_bounds__(256) void layernorm_kernel(
    const float* __restrict__ X, const float* __restrict__ g,
    const float* __restrict__ bta, float* __restrict__ outF,
    _Float16* __restrict__ outH) {
  int row  = blockIdx.x * 8 + (threadIdx.x >> 5);
  int lane = threadIdx.x & 31;
  const float* x = X + (size_t)row * DMODEL;
  float v[16];
  float s = 0.f;
  for (int i = 0; i < 16; ++i) { v[i] = x[lane + i * 32]; s += v[i]; }
  for (int o = 16; o; o >>= 1) s += __shfl_xor(s, o, 32);
  float mu = s * (1.f / DMODEL);
  float var = 0.f;
  for (int i = 0; i < 16; ++i) { float d = v[i] - mu; var += d * d; }
  for (int o = 16; o; o >>= 1) var += __shfl_xor(var, o, 32);
  float inv = rsqrtf(var * (1.f / DMODEL) + 1e-5f);
  for (int i = 0; i < 16; ++i) {
    int c = lane + i * 32;
    float y = (v[i] - mu) * inv * g[c] + bta[c];
    if (outF) outF[(size_t)row * DMODEL + c] = y;
    if (outH) outH[(size_t)row * DMODEL + c] = (_Float16)y;
  }
}

// ---------------------------------------------------------------------------
// WMMA GEMM: C[M,N] = A[M,K](f16) x W[K,N](f16) + bias, optional exact-GELU,
// optional f32 residual add; writes f32 and/or f16 outputs.
// Block tile 128x128, BK=32; 8 waves arranged 4(M) x 2(N); per-wave 32x64.
// A tile DMA'd by the TDM (wave 0 issues, others overlap B staging).
// ---------------------------------------------------------------------------
__global__ __launch_bounds__(256) void gemm_wmma_f16(
    const _Float16* __restrict__ A, const _Float16* __restrict__ W,
    const float* __restrict__ bias, const float* __restrict__ resid,
    float* __restrict__ outF, _Float16* __restrict__ outH,
    int M, int N, int K, int gelu) {
  (void)M;
  __shared__ _Float16 As[128 * 32];   // [m][k]
  __shared__ _Float16 Bs[128 * 32];   // [n][k]  (transposed on store)

  int tid  = threadIdx.x;
  int lane = tid & 31, wave = tid >> 5;
  int wm = wave >> 1, wn = wave & 1;
  int bm = blockIdx.y * 128, bn = blockIdx.x * 128;

  v8f acc[2][4] = {};

  int arow = tid >> 2;            // 0..63
  int acol = (tid & 3) * 8;       // 0,8,16,24
  int bkr  = tid >> 4;            // 0..15
  int bcol = (tid & 15) * 8;      // 0..120

  int mr = lane & 15;
  int ko8  = (lane >> 4) * 8;
  int ko16 = (lane >> 4) * 16;

  for (int k0 = 0; k0 < K; k0 += 32) {
    __syncthreads();
#if defined(HAVE_TDM)
    // TDM: 128x32 f16 A tile, row stride K elements, straight into LDS.
    if (tid < 32)
      tdm_load_tile_2d((unsigned int)(uintptr_t)&As[0],
                       A + (size_t)bm * K + k0,
                       /*tile_w=*/32, /*tile_h=*/128,
                       /*tensor_w=*/(unsigned int)K, /*tensor_h=*/128,
                       (unsigned long long)K);
#else
    for (int i = 0; i < 2; ++i) {
      int r = arow + i * 64;
      *(v4i*)(&As[r * 32 + acol]) =
          *(const v4i*)(A + (size_t)(bm + r) * K + k0 + acol);
    }
    if (k0 + 32 < K)
      __builtin_prefetch(A + (size_t)(bm + arow) * K + k0 + 32 + acol, 0, 1);
#endif
    // stage W tile transposed -> Bs[n][k]; issue both loads before stores
    {
      U8h d0, d1;
      d0.v = *(const v4i*)(W + (size_t)(k0 + bkr) * N + bn + bcol);
      d1.v = *(const v4i*)(W + (size_t)(k0 + bkr + 16) * N + bn + bcol);
      for (int j = 0; j < 8; ++j) Bs[(bcol + j) * 32 + bkr] = d0.h[j];
      for (int j = 0; j < 8; ++j) Bs[(bcol + j) * 32 + bkr + 16] = d1.h[j];
    }
    if (k0 + 32 < K)  // gfx1250 global_prefetch_b8 on next W k-tile
      __builtin_prefetch(W + (size_t)(k0 + 32 + bkr) * N + bn + bcol, 0, 1);
#if defined(HAVE_TDM)
    if (tid < 32) __builtin_amdgcn_s_wait_tensorcnt(0);
#endif
    __syncthreads();

    v16h af[2], bf[4];
    for (int t = 0; t < 2; ++t)
      af[t] = ldA_frag(&As[(wm * 32 + t * 16 + mr) * 32 + ko8]);
    for (int t = 0; t < 4; ++t)
      bf[t] = ldB_frag(&Bs[(wn * 64 + t * 16 + mr) * 32 + ko8]);

    for (int i = 0; i < 2; ++i)
      for (int j = 0; j < 4; ++j)
        acc[i][j] = wmma_f16(af[i], bf[j], acc[i][j]);
  }

  // epilogue: C layout — lane holds column (lane&15), rows mb..mb+7
  int mb = (lane >> 4) * 8;
  for (int i = 0; i < 2; ++i)
    for (int j = 0; j < 4; ++j) {
      int col = bn + wn * 64 + j * 16 + mr;
      float bv = bias ? bias[col] : 0.f;
      for (int r = 0; r < 8; ++r) {
        int row = bm + wm * 32 + i * 16 + mb + r;
        size_t o = (size_t)row * N + col;
        float vv = acc[i][j][r] + bv;
        if (gelu) vv = 0.5f * vv * (1.f + erff(vv * 0.70710678f));
        if (resid) vv += resid[o];
        if (outF) outF[o] = vv;
        if (outH) outH[o] = (_Float16)vv;
      }
    }
}

// ---------------------------------------------------------------------------
// Flash-style attention.  Block = (b, h, 64 q rows); 4 waves x 16 q rows.
// scores: S = (Q/8) Kt via WMMA (K rows are already B-fragment layout).
// Online softmax with shfl reductions in 16-lane halves; P goes through a
// per-wave LDS scratch to flip C-layout -> A-fragment; V staged transposed.
// ---------------------------------------------------------------------------
__global__ __launch_bounds__(128) void attention_kernel(
    const _Float16* __restrict__ Q, const _Float16* __restrict__ Kb,
    const _Float16* __restrict__ Vb, _Float16* __restrict__ Ctx) {
  int qblk = blockIdx.x, hh = blockIdx.y, b = blockIdx.z;
  int tid = threadIdx.x;
  int lane = tid & 31, wave = tid >> 5;
  int q0 = qblk * 64 + wave * 16;

  __shared__ _Float16 Vt[64][32];        // [d][key]
  __shared__ _Float16 Ps[4][16][32];     // per-wave P scratch [m][key]

  int mr = lane & 15;
  int ko8 = (lane >> 4) * 8;
  int mb  = (lane >> 4) * 8;

  // Q fragments, pre-scaled by 1/sqrt(64)
  v16h qf[2];
  for (int j = 0; j < 2; ++j) {
    const _Float16* p =
        Q + ((size_t)(b * LSEQ + q0 + mr)) * DMODEL + hh * DK + j * 32 + ko8;
    U16h u;
    u.i[0] = *(const v4i*)p;
    u.i[1] = *(const v4i*)(p + 16);
    for (int e = 0; e < 16; ++e) u.h[e] = (_Float16)((float)u.h[e] * 0.125f);
    qf[j] = u.v;
  }

  v8f acc[4] = {};
  float runmax[8], runsum[8];
  for (int r = 0; r < 8; ++r) { runmax[r] = -1e30f; runsum[r] = 0.f; }

  for (int kc = 0; kc < LSEQ / 32; ++kc) {
    int n0 = kc * 32;
    __syncthreads();
    // stage V chunk transposed: 128 threads x 16 halves
    {
      int key = tid >> 2;
      int dg  = (tid & 3) * 16;
      const _Float16* vp =
          Vb + ((size_t)(b * LSEQ + n0 + key)) * DMODEL + hh * DK + dg;
      U16h u;
      u.i[0] = *(const v4i*)vp;
      u.i[1] = *(const v4i*)(vp + 8);
      for (int j = 0; j < 16; ++j) Vt[dg + j][key] = u.h[j];
    }
    __syncthreads();

    // scores for 32 keys = two 16x16 tiles, K over d (two 32-steps)
    v8f s[2];
    for (int t = 0; t < 2; ++t) {
      v8f sv = {};
      for (int stp = 0; stp < 2; ++stp) {
        const _Float16* kp =
            Kb + ((size_t)(b * LSEQ + n0 + t * 16 + mr)) * DMODEL + hh * DK +
            stp * 32 + (lane >> 4) * 8;
        sv = wmma_f16(qf[stp], ldB_frag(kp), sv);
      }
      s[t] = sv;
    }

    // online softmax per row slot r (rows mb..mb+7 within this half)
    float p0[8], p1[8], scl[8];
    for (int r = 0; r < 8; ++r) {
      float m = fmaxf(s[0][r], s[1][r]);
      for (int o = 8; o; o >>= 1) m = fmaxf(m, __shfl_xor(m, o, 32));
      float nm = fmaxf(runmax[r], m);
      float sc = __expf(runmax[r] - nm);
      float a0 = __expf(s[0][r] - nm);
      float a1 = __expf(s[1][r] - nm);
      float rs = a0 + a1;
      for (int o = 8; o; o >>= 1) rs += __shfl_xor(rs, o, 32);
      runsum[r] = runsum[r] * sc + rs;
      runmax[r] = nm;
      p0[r] = a0; p1[r] = a1; scl[r] = sc;
    }
    for (int c = 0; c < 4; ++c)
      for (int r = 0; r < 8; ++r) acc[c][r] *= scl[r];

    // C-layout -> A-fragment via per-wave LDS (same-wave DS ops are in order)
    for (int r = 0; r < 8; ++r) {
      Ps[wave][mb + r][mr]      = (_Float16)p0[r];
      Ps[wave][mb + r][16 + mr] = (_Float16)p1[r];
    }
    v16h pf = ldA_frag(&Ps[wave][mr][ko8]);

    // O += P(16x32) x V(32x64)
    for (int c = 0; c < 4; ++c)
      acc[c] = wmma_f16(pf, ldB_frag(&Vt[c * 16 + mr][(lane >> 4) * 8]), acc[c]);
  }

  // normalize and store ctx (f16)
  for (int c = 0; c < 4; ++c)
    for (int r = 0; r < 8; ++r) {
      int row = q0 + mb + r;
      int col = c * 16 + mr;
      Ctx[((size_t)(b * LSEQ + row)) * DMODEL + hh * DK + col] =
          (_Float16)(acc[c][r] / runsum[r]);
    }
}

// ---------------------------------------------------------------------------
// final projection: out[row] = dot(hln[row,:], Wo[:,0]) + bo ; wave per row
// ---------------------------------------------------------------------------
__global__ __launch_bounds__(256) void outproj_kernel(
    const float* __restrict__ X, const float* __restrict__ Wo,
    const float* __restrict__ bo, float* __restrict__ out) {
  int row  = blockIdx.x * 8 + (threadIdx.x >> 5);
  int lane = threadIdx.x & 31;
  const float* x = X + (size_t)row * DMODEL;
  float s = 0.f;
  for (int i = 0; i < 16; ++i) {
    int c = lane + i * 32;
    s += x[c] * Wo[c];
  }
  for (int o = 16; o; o >>= 1) s += __shfl_xor(s, o, 32);
  if (lane == 0) out[row] = s + bo[0];
}

// ---------------------------------------------------------------------------
// host orchestration
// ---------------------------------------------------------------------------
extern "C" void kernel_launch(void* const* d_in, const int* in_sizes, int n_in,
                              void* d_out, int out_size, void* d_ws,
                              size_t ws_size, hipStream_t stream) {
  (void)in_sizes; (void)n_in; (void)out_size; (void)ws_size;

  // flatten order assumption: sorted dict keys (jax tree flatten)
  const float* X_obs   = (const float*)d_in[0];
  const float* maskp   = (const float*)d_in[1];
  const float* input_W = (const float*)d_in[2];
  const float* input_b = (const float*)d_in[3];
  // per-layer leaf offsets (sorted): b1,b2,bk,bo,bq,bv,ln1_b,ln1_g,ln2_b,ln2_g,w1,w2,wk,wo,wq,wv
  const float* out_W    = (const float*)d_in[68];
  const float* out_b    = (const float*)d_in[69];
  const float* out_ln_b = (const float*)d_in[70];
  const float* out_ln_g = (const float*)d_in[71];
  const float* pe       = (const float*)d_in[72];

  char* wsb = (char*)d_ws;
  size_t off = 0;
  auto alloc = [&](size_t bytes) -> char* {
    off = (off + 255) & ~(size_t)255;
    char* p = wsb + off;
    off += bytes;
    return p;
  };

  // f16 weights: per layer wq, wk, wv, wo, w1, w2
  _Float16* wf[4][6];
  for (int l = 0; l < 4; ++l) {
    for (int j = 0; j < 4; ++j) wf[l][j] = (_Float16*)alloc((size_t)512 * 512 * 2);
    wf[l][4] = (_Float16*)alloc((size_t)512 * 2048 * 2);
    wf[l][5] = (_Float16*)alloc((size_t)2048 * 512 * 2);
  }
  float*    h    = (float*)alloc((size_t)NROWS * DMODEL * 4);
  _Float16* hln  = (_Float16*)alloc((size_t)NROWS * DMODEL * 2);
  _Float16* qb   = (_Float16*)alloc((size_t)NROWS * DMODEL * 2);
  _Float16* kb   = (_Float16*)alloc((size_t)NROWS * DMODEL * 2);
  _Float16* vb   = (_Float16*)alloc((size_t)NROWS * DMODEL * 2);
  _Float16* ctx  = (_Float16*)alloc((size_t)NROWS * DMODEL * 2);
  _Float16* ffn  = (_Float16*)alloc((size_t)NROWS * DFF * 2);
  float*    pm   = (float*)alloc((size_t)LSEQ * 4);
  float*    lnf  = (float*)ffn;  // reuse: f32 16MB fits inside ffn's 32MB

  auto cvt = [&](const float* src, _Float16* dst, int n) {
    cvt_f32_f16<<<(n + 255) / 256, 256, 0, stream>>>(src, dst, n);
  };
  // weight conversion (sorted-leaf indices: w1=+10, w2=+11, wk=+12, wo=+13, wq=+14, wv=+15)
  for (int l = 0; l < 4; ++l) {
    int base = 4 + 16 * l;
    cvt((const float*)d_in[base + 14], wf[l][0], 512 * 512);    // wq
    cvt((const float*)d_in[base + 12], wf[l][1], 512 * 512);    // wk
    cvt((const float*)d_in[base + 15], wf[l][2], 512 * 512);    // wv
    cvt((const float*)d_in[base + 13], wf[l][3], 512 * 512);    // wo
    cvt((const float*)d_in[base + 10], wf[l][4], 512 * 2048);   // w1
    cvt((const float*)d_in[base + 11], wf[l][5], 2048 * 512);   // w2
  }

  posmean_kernel<<<(LSEQ + 255) / 256, 256, 0, stream>>>(X_obs, maskp, pm);
  embed_kernel<<<NROWS, 256, 0, stream>>>(X_obs, maskp, pm, input_W, input_b,
                                          pe, h);

  auto gemm = [&](const _Float16* A, const _Float16* W, const float* bias,
                  const float* resid, float* oF, _Float16* oH, int M, int N,
                  int K, int gelu) {
    dim3 grid(N / 128, M / 128);
    gemm_wmma_f16<<<grid, 256, 0, stream>>>(A, W, bias, resid, oF, oH, M, N, K,
                                            gelu);
  };

  for (int l = 0; l < 4; ++l) {
    int base = 4 + 16 * l;
    const float* b1    = (const float*)d_in[base + 0];
    const float* b2    = (const float*)d_in[base + 1];
    const float* bk_   = (const float*)d_in[base + 2];
    const float* bo_   = (const float*)d_in[base + 3];
    const float* bq_   = (const float*)d_in[base + 4];
    const float* bv_   = (const float*)d_in[base + 5];
    const float* ln1_b = (const float*)d_in[base + 6];
    const float* ln1_g = (const float*)d_in[base + 7];
    const float* ln2_b = (const float*)d_in[base + 8];
    const float* ln2_g = (const float*)d_in[base + 9];

    layernorm_kernel<<<NROWS / 8, 256, 0, stream>>>(h, ln1_g, ln1_b, nullptr,
                                                    hln);
    gemm(hln, wf[l][0], bq_, nullptr, nullptr, qb, NROWS, DMODEL, DMODEL, 0);
    gemm(hln, wf[l][1], bk_, nullptr, nullptr, kb, NROWS, DMODEL, DMODEL, 0);
    gemm(hln, wf[l][2], bv_, nullptr, nullptr, vb, NROWS, DMODEL, DMODEL, 0);

    attention_kernel<<<dim3(LSEQ / 64, NHEADS, BATCH), 128, 0, stream>>>(
        qb, kb, vb, ctx);

    // h = h + ctx @ wo + bo   (in-place residual)
    gemm(ctx, wf[l][3], bo_, h, h, nullptr, NROWS, DMODEL, DMODEL, 0);

    layernorm_kernel<<<NROWS / 8, 256, 0, stream>>>(h, ln2_g, ln2_b, nullptr,
                                                    hln);
    gemm(hln, wf[l][4], b1, nullptr, nullptr, ffn, NROWS, DFF, DMODEL, 1);
    gemm(ffn, wf[l][5], b2, h, h, nullptr, NROWS, DMODEL, DFF, 0);
  }

  layernorm_kernel<<<NROWS / 8, 256, 0, stream>>>(h, out_ln_g, out_ln_b, lnf,
                                                  nullptr);
  outproj_kernel<<<NROWS / 8, 256, 0, stream>>>(lnf, out_W, out_b,
                                                (float*)d_out);
}
